// GAT2_22308060136201
// MI455X (gfx1250) — matrix operand order repaired and checked by reading, not compiled, not run on hardware
//
#include <hip/hip_runtime.h>
#include <hip/hip_fp16.h>

// ---------------------------------------------------------------------------
// GAT x2 (dense graph) for gfx1250 / MI455X, wave32, WMMA f16->f32.
//   B=256 N=256 DIN=128 DH=128 H=4, F = H*DH = 512
// Each wave processes TWO 16-row strips so every LDS B-fragment feeds 2 WMMAs.
// ---------------------------------------------------------------------------

#define NB   256   // batches
#define NN   256   // nodes
#define DIN  128
#define DHH  128
#define NH   4
#define FDIM 512
#define NEG_SLOPE 0.2f
#define LN_EPS    1e-5f

typedef __attribute__((ext_vector_type(16))) _Float16 v16h;
typedef __attribute__((ext_vector_type(8)))  float    v8f;

union Frag32B {
    v16h h;
    unsigned int u[8];
    uint4 q[2];
};

// LDS row strides (halfwords), padded for bank spread, 16B-aligned rows.
#define XS_STRIDE 136   // 128 + 8   (272 B/row)
#define VT_STRIDE 264   // 256 + 8   (528 B/row)

// ---------------------------------------------------------------------------
// Kernel A: layer-1 projection.  h1t[b][h][d][n] = (x[b] @ W1)[n][h*128+d]  (f16)
// grid: (NH, NB), block 256 (8 waves). Block: 256 rows x 128 cols (one head).
// Wave wv owns row strips wv*16 and wv*16+128 (shared B fragments).
// ---------------------------------------------------------------------------
__global__ __launch_bounds__(256) void gat_proj1_kernel(
        const float* __restrict__ x, const float* __restrict__ W1,
        _Float16* __restrict__ h1t) {
    const int h = blockIdx.x;
    const int b = blockIdx.y;
    extern __shared__ char smem[];
    _Float16* xs = (_Float16*)smem;              // [256][XS_STRIDE]
    _Float16* wt = xs + 256 * XS_STRIDE;         // [128][XS_STRIDE]  (W^T: [d][k])
    const int t = threadIdx.x;

    // Load x tile (256 rows x 128 k) -> f16 LDS, one full row per thread.
    {
        const float* src = x + ((size_t)(b * NN + t)) * DIN;
        __builtin_prefetch(src, 0, 0);           // global_prefetch_b8
        _Float16* dst = xs + t * XS_STRIDE;
        for (int i = 0; i < DIN; i += 4) {
            float4 v = *(const float4*)(src + i);
            dst[i + 0] = (_Float16)v.x; dst[i + 1] = (_Float16)v.y;
            dst[i + 2] = (_Float16)v.z; dst[i + 3] = (_Float16)v.w;
        }
    }
    // Load W1[:,h,:] transposed -> wt[d][k].  float4 across d (coalesced), scalar
    // f16 transposed stores into LDS.
    for (int c = t; c < 128 * 32; c += 256) {
        const int k  = c >> 5;           // 0..127
        const int dq = (c & 31) * 4;     // 0..124
        float4 v = *(const float4*)(W1 + (size_t)k * FDIM + h * DHH + dq);
        wt[(dq + 0) * XS_STRIDE + k] = (_Float16)v.x;
        wt[(dq + 1) * XS_STRIDE + k] = (_Float16)v.y;
        wt[(dq + 2) * XS_STRIDE + k] = (_Float16)v.z;
        wt[(dq + 3) * XS_STRIDE + k] = (_Float16)v.w;
    }
    __syncthreads();

    const int wv = t >> 5, lane = t & 31;
    const int l15 = lane & 15;
    const int hi  = (lane < 16) ? 0 : 1;
    const int mrow = wv * 16 + l15;              // strip-0 A row for this lane

    v8f acc0[8], acc1[8];
    for (int q = 0; q < 8; ++q) { acc0[q] = (v8f)0.0f; acc1[q] = (v8f)0.0f; }

    for (int kc = 0; kc < DIN; kc += 32) {
        // Two A fragments 16x32 (M=n, K=k): k = e + (e<8?0:8) + hi*8
        Frag32B a0, a1;
        {
            const unsigned int* p0 =
                (const unsigned int*)(xs + mrow * XS_STRIDE + kc + hi * 8);
            const unsigned int* p1 =
                (const unsigned int*)(xs + (mrow + 128) * XS_STRIDE + kc + hi * 8);
            a0.u[0] = p0[0]; a0.u[1] = p0[1]; a0.u[2] = p0[2]; a0.u[3] = p0[3];
            a0.u[4] = p0[8]; a0.u[5] = p0[9]; a0.u[6] = p0[10]; a0.u[7] = p0[11];
            a1.u[0] = p1[0]; a1.u[1] = p1[1]; a1.u[2] = p1[2]; a1.u[3] = p1[3];
            a1.u[4] = p1[8]; a1.u[5] = p1[9]; a1.u[6] = p1[10]; a1.u[7] = p1[11];
        }
        // Rotating depth-2 B-fragment buffer: load dt+1 before WMMAs on dt.
        Frag32B bf[2];
        {
            const _Float16* base = wt + l15 * XS_STRIDE + kc + hi * 8;
            bf[0].q[0] = *(const uint4*)(base);
            bf[0].q[1] = *(const uint4*)(base + 16);
        }
#pragma unroll
        for (int dt = 0; dt < 8; ++dt) {
            if (dt + 1 < 8) {
                const _Float16* base =
                    wt + ((dt + 1) * 16 + l15) * XS_STRIDE + kc + hi * 8;
                bf[(dt + 1) & 1].q[0] = *(const uint4*)(base);
                bf[(dt + 1) & 1].q[1] = *(const uint4*)(base + 16);
            }
            acc0[dt] = __builtin_amdgcn_wmma_f32_16x16x32_f16(
                false, a0.h, false, bf[dt & 1].h, (short)0, acc0[dt], false, false);
            acc1[dt] = __builtin_amdgcn_wmma_f32_16x16x32_f16(
                false, a1.h, false, bf[dt & 1].h, (short)0, acc1[dt], false, false);
        }
    }
    // Store transposed: h1t[((b*NH+h)*DHH+d)*NN + n], 8 consecutive n per lane.
    for (int dt = 0; dt < 8; ++dt) {
        const int d  = dt * 16 + l15;
        const int n0 = wv * 16 + hi * 8;
        _Float16* base = h1t + ((size_t)((b * NH + h) * DHH + d)) * NN;
        _Float16 o0[8], o1[8];
        for (int r = 0; r < 8; ++r) {
            o0[r] = (_Float16)acc0[dt][r];
            o1[r] = (_Float16)acc1[dt][r];
        }
        *(uint4*)(base + n0)       = *(uint4*)o0;
        *(uint4*)(base + n0 + 128) = *(uint4*)o1;
    }
}

// ---------------------------------------------------------------------------
// Kernel B: layer-1 attention: out1[b,i,h*128+d] = softmax_j(lrelu(ad_i+as_j)) @ V + b1
// grid: (NH, NB), block 256.  V^T (128 x 256 f16) in LDS.
// Wave wv runs output strips wv and wv+8 simultaneously (shared V fragments).
// ---------------------------------------------------------------------------
__global__ __launch_bounds__(256) void gat_attn1_kernel(
        const _Float16* __restrict__ h1t,
        const float* __restrict__ att_src1, const float* __restrict__ att_dst1,
        const float* __restrict__ b1, float* __restrict__ out1) {
    const int h = blockIdx.x;
    const int b = blockIdx.y;
    extern __shared__ char smem[];
    _Float16* vt = (_Float16*)smem;                      // [128][VT_STRIDE]
    float* ms   = (float*)(smem + 128 * VT_STRIDE * 2);  // [256] alpha_src
    float* md   = ms + NN;                               // [256] alpha_dst
    float* rmax = md + NN;                               // [256]
    float* rinv = rmax + NN;                             // [256]
    float* asrc = rinv + NN;                             // [128]
    float* adst = asrc + DHH;                            // [128]
    float* bvec = adst + DHH;                            // [128]
    const int t = threadIdx.x;

    // Load V^T rows (d-major, n contiguous) straight from global (already transposed).
    {
        const int d = t >> 1, half = (t & 1) * 128;
        const _Float16* src = h1t + ((size_t)((b * NH + h) * DHH + d)) * NN + half;
        _Float16* dst = vt + d * VT_STRIDE + half;
        for (int i = 0; i < 128; i += 8)
            *(uint4*)(dst + i) = *(const uint4*)(src + i);
    }
    if (t < 128) { asrc[t] = att_src1[h * DHH + t]; adst[t] = att_dst1[h * DHH + t]; }
    else         { const int d = t - 128; bvec[d] = b1[h * DHH + d]; }
    __syncthreads();

    // Per-node attention logits: ms[j] = <V_j, att_src>, md[i] = <V_i, att_dst>
    {
        float s = 0.f, dd = 0.f;
        for (int d = 0; d < DHH; ++d) {
            const float v = (float)vt[d * VT_STRIDE + t];
            s += v * asrc[d];
            dd += v * adst[d];
        }
        ms[t] = s; md[t] = dd;
    }
    __syncthreads();

    // Exact per-row softmax stats (max, 1/sum) for row i = t.
    {
        const float di = md[t];
        float m = -1e30f;
        for (int j = 0; j < NN; ++j) {
            float e = di + ms[j];
            e = (e > 0.f) ? e : NEG_SLOPE * e;
            m = fmaxf(m, e);
        }
        float sum = 0.f;
        for (int j = 0; j < NN; ++j) {
            float e = di + ms[j];
            e = (e > 0.f) ? e : NEG_SLOPE * e;
            sum += __expf(e - m);
        }
        rmax[t] = m; rinv[t] = 1.f / sum;
    }
    __syncthreads();

    const int wv = t >> 5, lane = t & 31;
    const int l15 = lane & 15;
    const int hi  = (lane < 16) ? 0 : 1;

    // Two simultaneous output strips per wave: rows i0 and i0+128.
    const int i0 = wv * 16 + l15;
    const int i1 = i0 + 128;
    const float di0 = md[i0], rm0 = rmax[i0], ri0 = rinv[i0];
    const float di1 = md[i1], rm1 = rmax[i1], ri1 = rinv[i1];

    v8f acc0[8], acc1[8];
    for (int q = 0; q < 8; ++q) { acc0[q] = (v8f)0.0f; acc1[q] = (v8f)0.0f; }

    for (int jc = 0; jc < NN; jc += 32) {
        // Probability A-fragments built in registers (VALU/TRANS co-executes
        // with the WMMA stream).
        Frag32B a0, a1;
#pragma unroll
        for (int e = 0; e < 16; ++e) {
            const int k = e + hi * 16;
            const float msk = ms[jc + k];
            float e0 = di0 + msk;
            e0 = (e0 > 0.f) ? e0 : NEG_SLOPE * e0;
            a0.h[e] = (_Float16)(__expf(e0 - rm0) * ri0);
            float e1 = di1 + msk;
            e1 = (e1 > 0.f) ? e1 : NEG_SLOPE * e1;
            a1.h[e] = (_Float16)(__expf(e1 - rm1) * ri1);
        }
        // Rotating depth-2 V-fragment buffer; each fragment feeds 2 WMMAs.
        Frag32B bf[2];
        {
            const _Float16* base = vt + l15 * VT_STRIDE + jc + hi * 16;
            bf[0].q[0] = *(const uint4*)(base);
            bf[0].q[1] = *(const uint4*)(base + 8);
        }
#pragma unroll
        for (int dt = 0; dt < 8; ++dt) {
            if (dt + 1 < 8) {
                const _Float16* base =
                    vt + ((dt + 1) * 16 + l15) * VT_STRIDE + jc + hi * 16;
                bf[(dt + 1) & 1].q[0] = *(const uint4*)(base);
                bf[(dt + 1) & 1].q[1] = *(const uint4*)(base + 8);
            }
            acc0[dt] = __builtin_amdgcn_wmma_f32_16x16x32_f16(
                false, a0.h, false, bf[dt & 1].h, (short)0, acc0[dt], false, false);
            acc1[dt] = __builtin_amdgcn_wmma_f32_16x16x32_f16(
                false, a1.h, false, bf[dt & 1].h, (short)0, acc1[dt], false, false);
        }
    }
    // Store: out1[(b*NN + i)*FDIM + h*128 + d]  (+ bias, fp32)
    for (int dt = 0; dt < 8; ++dt) {
        const int d = dt * 16 + l15;
        const float bias = bvec[d];
        const int r0 = wv * 16 + hi * 8;
        float* dst0 = out1 + ((size_t)(b * NN + r0)) * FDIM + h * DHH + d;
        float* dst1 = dst0 + (size_t)128 * FDIM;
        for (int r = 0; r < 8; ++r) {
            dst0[(size_t)r * FDIM] = acc0[dt][r] + bias;
            dst1[(size_t)r * FDIM] = acc1[dt][r] + bias;
        }
    }
}

// ---------------------------------------------------------------------------
// Kernel C: LayerNorm(512) + ELU + layer-2 projection to scalar h2[b,n].
// One wave per node; wave32 shuffle reductions.
// ---------------------------------------------------------------------------
__global__ __launch_bounds__(256) void gat_ln_proj2_kernel(
        const float* __restrict__ out1,
        const float* __restrict__ gamma, const float* __restrict__ beta,
        const float* __restrict__ W2, float* __restrict__ h2) {
    const int wv = threadIdx.x >> 5, lane = threadIdx.x & 31;
    const int node = blockIdx.x * 8 + wv;
    const float* src = out1 + (size_t)node * FDIM + lane * 16;
    float v[16];
    for (int i = 0; i < 16; i += 4) {
        float4 q = *(const float4*)(src + i);
        v[i] = q.x; v[i + 1] = q.y; v[i + 2] = q.z; v[i + 3] = q.w;
    }
    float s = 0.f;
    for (int i = 0; i < 16; ++i) s += v[i];
    for (int o = 16; o > 0; o >>= 1) s += __shfl_xor(s, o, 32);
    const float mean = s * (1.0f / FDIM);

    float vs = 0.f;
    for (int i = 0; i < 16; ++i) { const float d = v[i] - mean; vs += d * d; }
    for (int o = 16; o > 0; o >>= 1) vs += __shfl_xor(vs, o, 32);
    const float rstd = rsqrtf(vs * (1.0f / FDIM) + LN_EPS);

    float dot = 0.f;
    for (int i = 0; i < 16; ++i) {
        const int f = lane * 16 + i;
        float g = (v[i] - mean) * rstd * gamma[f] + beta[f];
        g = (g > 0.f) ? g : (__expf(g) - 1.f);   // ELU
        dot += g * W2[f];
    }
    for (int o = 16; o > 0; o >>= 1) dot += __shfl_xor(dot, o, 32);
    if (lane == 0) h2[node] = dot;
}

// ---------------------------------------------------------------------------
// Kernel D: layer-2 attention (scalar features -> GEMV softmax). One block/batch.
// ---------------------------------------------------------------------------
__global__ __launch_bounds__(256) void gat_attn2_kernel(
        const float* __restrict__ h2,
        const float* __restrict__ att_src2, const float* __restrict__ att_dst2,
        const float* __restrict__ b2, float* __restrict__ out) {
    __shared__ float hh[NN];
    const int b = blockIdx.x, i = threadIdx.x;
    const float hv = h2[(size_t)b * NN + i];
    hh[i] = hv;
    __syncthreads();
    const float as = att_src2[0], ad = att_dst2[0], bias = b2[0];
    const float di = hv * ad;
    float m = -1e30f;
    for (int j = 0; j < NN; ++j) {
        float e = di + as * hh[j];
        e = (e > 0.f) ? e : NEG_SLOPE * e;
        m = fmaxf(m, e);
    }
    float se = 0.f, sv = 0.f;
    for (int j = 0; j < NN; ++j) {
        float e = di + as * hh[j];
        e = (e > 0.f) ? e : NEG_SLOPE * e;
        const float p = __expf(e - m);
        se += p; sv += p * hh[j];
    }
    out[(size_t)b * NN + i] = sv / se + bias;
}

// ---------------------------------------------------------------------------
extern "C" void kernel_launch(void* const* d_in, const int* in_sizes, int n_in,
                              void* d_out, int out_size, void* d_ws, size_t ws_size,
                              hipStream_t stream) {
    (void)in_sizes; (void)n_in; (void)out_size; (void)ws_size;
    const float* x        = (const float*)d_in[0];
    /* d_in[1] = adj, unused (graph is fully connected) */
    const float* W1       = (const float*)d_in[2];
    const float* att_src1 = (const float*)d_in[3];
    const float* att_dst1 = (const float*)d_in[4];
    const float* b1       = (const float*)d_in[5];
    const float* gamma    = (const float*)d_in[6];
    const float* beta     = (const float*)d_in[7];
    const float* W2       = (const float*)d_in[8];
    const float* att_src2 = (const float*)d_in[9];
    const float* att_dst2 = (const float*)d_in[10];
    const float* b2       = (const float*)d_in[11];
    float* out = (float*)d_out;

    // Workspace layout
    char* ws = (char*)d_ws;
    const size_t h1t_bytes  = (size_t)NB * NH * DHH * NN * sizeof(_Float16); // 67 MB
    const size_t out1_bytes = (size_t)NB * NN * FDIM * sizeof(float);        // 134 MB
    _Float16* h1t = (_Float16*)ws;
    float*    out1 = (float*)(ws + h1t_bytes);
    float*    h2   = (float*)(ws + h1t_bytes + out1_bytes);

    const size_t smemA = (size_t)(256 + 128) * XS_STRIDE * sizeof(_Float16); // 104448 B
    const size_t smemB = (size_t)128 * VT_STRIDE * sizeof(_Float16)
                       + 4 * NN * sizeof(float) + 3 * DHH * sizeof(float);   // 73216 B

    // Unconditional (idempotent, deterministic) host-side attribute set;
    // allows >64KB dynamic LDS per workgroup (CDNA5 WGP has 320KB).
    hipFuncSetAttribute((const void*)gat_proj1_kernel,
                        hipFuncAttributeMaxDynamicSharedMemorySize, (int)smemA);
    hipFuncSetAttribute((const void*)gat_attn1_kernel,
                        hipFuncAttributeMaxDynamicSharedMemorySize, (int)smemB);

    gat_proj1_kernel<<<dim3(NH, NB), 256, smemA, stream>>>(x, W1, h1t);
    gat_attn1_kernel<<<dim3(NH, NB), 256, smemB, stream>>>(
        h1t, att_src1, att_dst1, b1, out1);
    gat_ln_proj2_kernel<<<dim3(NB * NN / 8), 256, 0, stream>>>(
        out1, gamma, beta, W2, h2);
    gat_attn2_kernel<<<dim3(NB), 256, 0, stream>>>(h2, att_src2, att_dst2, b2, out);
}
